// HGTAttention_32349693674122
// MI455X (gfx1250) — compile-verified, matrix operand order checked by reading, not run on hardware
//
#include <hip/hip_runtime.h>

#define D  128
#define H  8
#define DK 16

typedef __bf16 bf16;
typedef __attribute__((ext_vector_type(4)))  bf16  v4bf;
typedef __attribute__((ext_vector_type(8)))  bf16  v8bf;
typedef __attribute__((ext_vector_type(16))) bf16  v16bf;
typedef __attribute__((ext_vector_type(8)))  float v8f;

// ---------------------------------------------------------------- utilities
__device__ __forceinline__ v16bf load16(const bf16* p) {
  // lane fetches two contiguous 8-element (16B) chunks 16 elements apart,
  // matching the 16-bit A/B fragment layout (K 0..7 & 16..23 or 8..15 & 24..31)
  v8bf lo = *(const v8bf*)(p);
  v8bf hi = *(const v8bf*)(p + 16);
  return __builtin_shufflevector(lo, hi, 0,1,2,3,4,5,6,7,8,9,10,11,12,13,14,15);
}

__device__ __forceinline__ void atomicMaxF(float* a, float v) {
  // classic two-sided trick: works with init = -inf
  if (v >= 0.f) atomicMax((int*)a, __float_as_int(v));
  else          atomicMin((unsigned int*)a, __float_as_uint(v));
}

// ---------------------------------------------------------------- prep
__global__ void cvt_feat(const float* __restrict__ f, bf16* __restrict__ fb, int n4) {
  int i = blockIdx.x * blockDim.x + threadIdx.x;
  if (i >= n4) return;
  float4 v = ((const float4*)f)[i];
  v4bf o; o[0] = (bf16)v.x; o[1] = (bf16)v.y; o[2] = (bf16)v.z; o[3] = (bf16)v.w;
  *(v4bf*)(fb + (size_t)i * 4) = o;
}

// Fold per-head relation transforms into projection weights.
// B matrices stored column-major (row c = output column, D entries over k).
__global__ void prep_w(const float* __restrict__ Wq, const float* __restrict__ Wk,
                       const float* __restrict__ Wv, const float* __restrict__ watt,
                       const float* __restrict__ wmsg,
                       bf16* __restrict__ Bq, bf16* __restrict__ Bk, bf16* __restrict__ Bv) {
  int gid = blockIdx.x * blockDim.x + threadIdx.x;
  if (gid >= 3 * D * D) return;
  int mat = gid / (D * D);
  int idx = gid % (D * D);
  int c = idx / D;   // output column
  int d = idx % D;   // k index
  if (mat == 0) { Bq[idx] = (bf16)Wq[c * D + d]; return; }
  const float* W = (mat == 1) ? Wk : Wv;
  const float* T = (mat == 1) ? watt : wmsg;
  int h = c / DK, j = c % DK;
  float s = 0.f;
  #pragma unroll
  for (int i = 0; i < DK; ++i)
    s += W[(h * DK + i) * D + d] * T[(h * DK + i) * DK + j];
  ((mat == 1) ? Bk : Bv)[idx] = (bf16)s;
}

__global__ void prep_b(const float* __restrict__ bk, const float* __restrict__ bv,
                       const float* __restrict__ watt, const float* __restrict__ wmsg,
                       float* __restrict__ bke, float* __restrict__ bve) {
  int t = threadIdx.x;
  if (t >= 2 * D) return;
  const float* b = (t < D) ? bk : bv;
  const float* T = (t < D) ? watt : wmsg;
  int c = t % D, h = c / DK, j = c % DK;
  float s = 0.f;
  #pragma unroll
  for (int i = 0; i < DK; ++i)
    s += b[h * DK + i] * T[(h * DK + i) * DK + j];
  ((t < D) ? bke : bve)[c] = s;
}

__global__ void init_buf(float* __restrict__ out, float* __restrict__ m,
                         float* __restrict__ denom, int N) {
  int i = blockIdx.x * blockDim.x + threadIdx.x;
  if (i < N * D) out[i] = 0.f;
  if (i < N * H) { m[i] = -__builtin_inff(); denom[i] = 0.f; }
}

// ---------------------------------------------------------------- projection GEMM
// Block = 256 = 8 waves. Block owns a 16-node tile; wave w owns columns
// [16w,16w+16) of q, k', v' (3 accumulators). K swept 0..128 in steps of 32.
__global__ void __launch_bounds__(256)
proj(const bf16* __restrict__ featb,
     const bf16* __restrict__ Bq, const bf16* __restrict__ Bk, const bf16* __restrict__ Bv,
     const float* __restrict__ bq, const float* __restrict__ bke, const float* __restrict__ bve,
     float* __restrict__ q, float* __restrict__ k2, float* __restrict__ v2, int N) {
  const int wave = threadIdx.x >> 5;
  const int lane = threadIdx.x & 31;
  const int half = lane >> 4;
  const int l16  = lane & 15;
  const int row0 = blockIdx.x * 16;
  int arow = row0 + l16; if (arow >= N) arow = N - 1;   // clamp load, guard store
  const int col = wave * 16 + l16;

  v8f cq = {}, ck = {}, cv = {};
  #pragma unroll
  for (int kb = 0; kb < D; kb += 32) {
    const bf16* ap = featb + (size_t)arow * D + kb + half * 8;
    v16bf a = load16(ap);
    const size_t bo = (size_t)col * D + kb + half * 8;
    v16bf fq = load16(Bq + bo);
    v16bf fk = load16(Bk + bo);
    v16bf fv = load16(Bv + bo);
    cq = __builtin_amdgcn_wmma_f32_16x16x32_bf16(false, a, false, fq, (short)0, cq, false, false);
    ck = __builtin_amdgcn_wmma_f32_16x16x32_bf16(false, a, false, fk, (short)0, ck, false, false);
    cv = __builtin_amdgcn_wmma_f32_16x16x32_bf16(false, a, false, fv, (short)0, cv, false, false);
  }
  const float aq = bq[col], ak = bke[col], av = bve[col];
  #pragma unroll
  for (int i = 0; i < 8; ++i) {
    int r = row0 + half * 8 + i;
    if (r < N) {
      size_t o = (size_t)r * D + col;
      q[o]  = cq[i] + aq;
      k2[o] = ck[i] + ak;
      v2[o] = cv[i] + av;
    }
  }
}

// ---------------------------------------------------------------- edge phases
__global__ void __launch_bounds__(256)
edge_logits(const float* __restrict__ q, const float* __restrict__ k2,
            const int* __restrict__ src, const int* __restrict__ dst,
            const float* __restrict__ mu, float* __restrict__ logits,
            float* __restrict__ m, int E) {
  int e = blockIdx.x * blockDim.x + threadIdx.x;
  if (e >= E) return;
  int s = src[e], d = dst[e];
  const float4* qp = (const float4*)(q  + (size_t)d * D);
  const float4* kp = (const float4*)(k2 + (size_t)s * D);
  float acc[H] = {};
  #pragma unroll
  for (int i = 0; i < D / 4; ++i) {
    float4 qa = qp[i], ka = kp[i];
    acc[i >> 2] += qa.x * ka.x + qa.y * ka.y + qa.z * ka.z + qa.w * ka.w;
  }
  float scale = mu[0] * 0.25f;   // mu / sqrt(dk), dk = 16
  #pragma unroll
  for (int h = 0; h < H; ++h) {
    float lg = acc[h] * scale;
    logits[(size_t)e * H + h] = lg;
    atomicMaxF(&m[(size_t)d * H + h], lg);
  }
}

__global__ void __launch_bounds__(256)
edge_exp(const int* __restrict__ dst, const float* __restrict__ m,
         float* __restrict__ logits, float* __restrict__ denom, int E) {
  int e = blockIdx.x * blockDim.x + threadIdx.x;
  if (e >= E) return;
  int d = dst[e];
  #pragma unroll
  for (int h = 0; h < H; ++h) {
    float ex = __expf(logits[(size_t)e * H + h] - m[(size_t)d * H + h]);
    logits[(size_t)e * H + h] = ex;
    atomicAdd(&denom[(size_t)d * H + h], ex);
  }
}

__global__ void __launch_bounds__(256)
edge_aggr(const int* __restrict__ src, const int* __restrict__ dst,
          const float* __restrict__ logits, const float* __restrict__ denom,
          const float* __restrict__ v2, float* __restrict__ out, int E) {
  int e = blockIdx.x * blockDim.x + threadIdx.x;
  if (e >= E) return;
  int s = src[e], d = dst[e];
  float at[H];
  #pragma unroll
  for (int h = 0; h < H; ++h)
    at[h] = logits[(size_t)e * H + h] / denom[(size_t)d * H + h];
  const float4* vp = (const float4*)(v2 + (size_t)s * D);
  float* op = out + (size_t)d * D;
  #pragma unroll
  for (int i = 0; i < D / 4; ++i) {
    float4 vv = vp[i];
    float a = at[i >> 2];
    atomicAdd(op + i * 4 + 0, vv.x * a);
    atomicAdd(op + i * 4 + 1, vv.y * a);
    atomicAdd(op + i * 4 + 2, vv.z * a);
    atomicAdd(op + i * 4 + 3, vv.w * a);
  }
}

// ---------------------------------------------------------------- launch
extern "C" void kernel_launch(void* const* d_in, const int* in_sizes, int n_in,
                              void* d_out, int out_size, void* d_ws, size_t ws_size,
                              hipStream_t stream) {
  const float* feat = (const float*)d_in[0];
  const float* Wk   = (const float*)d_in[1];
  const float* bk   = (const float*)d_in[2];
  const float* Wq   = (const float*)d_in[3];
  const float* bq   = (const float*)d_in[4];
  const float* Wv   = (const float*)d_in[5];
  const float* bv   = (const float*)d_in[6];
  const float* watt = (const float*)d_in[7];
  const float* wmsg = (const float*)d_in[8];
  const float* mu   = (const float*)d_in[9];
  const int*   src  = (const int*)d_in[10];
  const int*   dst  = (const int*)d_in[11];
  float* out = (float*)d_out;

  const int N = in_sizes[0] / D;
  const int E = in_sizes[10];

  // workspace carving (256B aligned)
  char* w = (char*)d_ws;
  auto carve = [&](size_t bytes) { char* p = w; w += (bytes + 255) & ~(size_t)255; return p; };
  bf16*  featb  = (bf16*)carve((size_t)N * D * sizeof(bf16));
  bf16*  Bq     = (bf16*)carve((size_t)D * D * sizeof(bf16));
  bf16*  Bk     = (bf16*)carve((size_t)D * D * sizeof(bf16));
  bf16*  Bv     = (bf16*)carve((size_t)D * D * sizeof(bf16));
  float* bke    = (float*)carve(D * sizeof(float));
  float* bve    = (float*)carve(D * sizeof(float));
  float* qbuf   = (float*)carve((size_t)N * D * sizeof(float));
  float* k2buf  = (float*)carve((size_t)N * D * sizeof(float));
  float* v2buf  = (float*)carve((size_t)N * D * sizeof(float));
  float* logits = (float*)carve((size_t)E * H * sizeof(float));
  float* mbuf   = (float*)carve((size_t)N * H * sizeof(float));
  float* dbuf   = (float*)carve((size_t)N * H * sizeof(float));

  const int n4 = N * D / 4;
  cvt_feat<<<(n4 + 255) / 256, 256, 0, stream>>>(feat, featb, n4);
  prep_w<<<(3 * D * D + 255) / 256, 256, 0, stream>>>(Wq, Wk, Wv, watt, wmsg, Bq, Bk, Bv);
  prep_b<<<1, 256, 0, stream>>>(bk, bv, watt, wmsg, bke, bve);
  init_buf<<<((size_t)N * D + 255) / 256, 256, 0, stream>>>(out, mbuf, dbuf, N);

  proj<<<(N + 15) / 16, 256, 0, stream>>>(featb, Bq, Bk, Bv, bq, bke, bve,
                                          qbuf, k2buf, v2buf, N);

  const int eb = (E + 255) / 256;
  edge_logits<<<eb, 256, 0, stream>>>(qbuf, k2buf, src, dst, mu, logits, mbuf, E);
  edge_exp<<<eb, 256, 0, stream>>>(dst, mbuf, logits, dbuf, E);
  edge_aggr<<<eb, 256, 0, stream>>>(src, dst, logits, dbuf, v2buf, out, E);
}
